// IntegratedGraphConvLayer_70557722738855
// MI455X (gfx1250) — compile-verified, matrix operand order checked by reading, not run on hardware
//
#include <hip/hip_runtime.h>

// ---------------------------------------------------------------------------
// MI455X / gfx1250, wave32. All GEMMs via v_wmma_f32_16x16x32_f16.
// Memory-bound problem (~420MB @ 23.3TB/s ~= 18us floor); we avoid the 102MB
// per-edge value materialization by computing scatter winners first and only
// writing winning rows.
// ---------------------------------------------------------------------------

typedef __attribute__((ext_vector_type(16))) _Float16 v16h;
typedef __attribute__((ext_vector_type(8)))  _Float16 v8h;
typedef __attribute__((ext_vector_type(8)))  float    v8f;

#define DIN    128
#define DEDGE  172
#define DTIME  100
#define KMSG_R 528           // real message input dim
#define KMSG   544           // padded to multiple of 32 (17 k-chunks)
#define KUPD   256           // update input dim (8 k-chunks)

// ---- WMMA fragment load -----------------------------------------------------
// 16-bit A operand layout (ISA 7.12.2): lane L holds row M = L&15; lanes 0-15
// carry K = [kBase+0..7] and [kBase+16..23], lanes 16-31 carry K = [kBase+8..15]
// and [kBase+24..31]. B operand (stored as Wt[n][k]) has the identical per-lane
// pattern with N = L&15. So one loader serves both: two 16-byte chunks.
__device__ __forceinline__ v16h frag16(const _Float16* __restrict__ row,
                                       int kBase, int half) {
  const v8h lo = *(const v8h*)(row + kBase + 8 * half);
  const v8h hi = *(const v8h*)(row + kBase + 8 * half + 16);
  return __builtin_shufflevector(lo, hi, 0,1,2,3,4,5,6,7,8,9,10,11,12,13,14,15);
}

__device__ __forceinline__ v8f wmma16(v16h a, v16h b, v8f c) {
  // 8 args: (neg_a, A, neg_b, B, c_mod, C, reuse_a, reuse_b)
  return __builtin_amdgcn_wmma_f32_16x16x32_f16(false, a, false, b,
                                                (short)0, c, false, false);
}

// ---- Kernel 1: weights -> f16, transposed [n][k], msg_W1 K-padded to 544 ----
__global__ void prep_weights(const float* __restrict__ msgW1,
                             const float* __restrict__ msgW2,
                             const float* __restrict__ updW1,
                             const float* __restrict__ updW2,
                             _Float16* __restrict__ w1t,
                             _Float16* __restrict__ w2t,
                             _Float16* __restrict__ u1t,
                             _Float16* __restrict__ u2t) {
  const int tid    = blockIdx.x * blockDim.x + threadIdx.x;
  const int stride = gridDim.x * blockDim.x;
  for (int i = tid; i < 128 * KMSG; i += stride) {
    int n = i / KMSG, k = i % KMSG;
    w1t[i] = (k < KMSG_R) ? (_Float16)msgW1[k * 128 + n] : (_Float16)0.0f;
  }
  for (int i = tid; i < 128 * 128; i += stride) {
    int n = i / 128, k = i % 128;
    w2t[i] = (_Float16)msgW2[k * 128 + n];
  }
  for (int i = tid; i < 128 * KUPD; i += stride) {
    int n = i / KUPD, k = i % KUPD;
    u1t[i] = (_Float16)updW1[k * 128 + n];
  }
  for (int i = tid; i < 128 * 128; i += stride) {
    int n = i / 128, k = i % 128;
    u2t[i] = (_Float16)updW2[k * 128 + n];
  }
}

// ---- Kernel 2/3: scatter-winner (segment_max of position) -------------------
__global__ void winner_init(int* __restrict__ winner, int n) {
  int i = blockIdx.x * blockDim.x + threadIdx.x;
  if (i < n) winner[i] = -1;
}

__global__ void winner_scatter(const int* __restrict__ srcIds,
                               const int* __restrict__ dstIds,
                               int* __restrict__ winner, int nEdges) {
  int e = blockIdx.x * blockDim.x + threadIdx.x;
  if (e < nEdges) {
    atomicMax(&winner[dstIds[e]], 2 * e);       // dst written first -> pos 2e
    atomicMax(&winner[srcIds[e]], 2 * e + 1);   // then src          -> pos 2e+1
  }
}

// ---- Kernel 4: identity fallback copy ---------------------------------------
__global__ void copy_base(const float4* __restrict__ emb,
                          float4* __restrict__ out, int n4) {
  int i = blockIdx.x * blockDim.x + threadIdx.x;
  if (i < n4) out[i] = emb[i];
}

// ---- Kernel 5: per-16-edge-tile fused MLPs + guarded winning writes ---------
__global__ __launch_bounds__(256) void edge_tile_kernel(
    const float* __restrict__ emb,
    const int*   __restrict__ srcIds,
    const int*   __restrict__ dstIds,
    const float* __restrict__ edgeF,
    const float* __restrict__ timeF,
    const _Float16* __restrict__ W1t, const float* __restrict__ b1,
    const _Float16* __restrict__ W2t, const float* __restrict__ b2,
    const _Float16* __restrict__ U1t, const float* __restrict__ ub1,
    const _Float16* __restrict__ U2t, const float* __restrict__ ub2,
    const int* __restrict__ winner,
    float* __restrict__ out, int nEdges) {

  __shared__ __attribute__((aligned(16))) _Float16 Am[16 * KMSG];  // msg input
  __shared__ __attribute__((aligned(16))) _Float16 Hs[16 * 128];   // hidden
  __shared__ __attribute__((aligned(16))) _Float16 Ms[16 * 128];   // message
  __shared__ int sSrc[16], sDst[16];

  const int tid  = threadIdx.x;
  const int wave = tid >> 5;           // 0..7 -> 16 output columns each
  const int lane = tid & 31;
  const int col  = lane & 15;          // M for A-frag rows, N for B-frag rows
  const int half = lane >> 4;          // K-half selector per ISA layout
  const int e0   = blockIdx.x * 16;
  const int nCol = wave * 16 + col;    // this lane's output column (0..127)

  if (tid < 16) {
    int e = e0 + tid;
    sSrc[tid] = (e < nEdges) ? srcIds[e] : 0;
    sDst[tid] = (e < nEdges) ? dstIds[e] : 0;
  }
  __builtin_prefetch(W1t + nCol * KMSG, 0, 3);   // global_prefetch_b8
  __builtin_prefetch(U1t + nCol * KUPD, 0, 3);
  __syncthreads();

  // Stage message input tile [16 x 544] f16 into LDS:
  // [src_emb | dst_emb | edge_features | time_features | zero-pad]
  for (int i = tid; i < 16 * KMSG; i += 256) {
    int m = i / KMSG, c = i % KMSG;
    int e = e0 + m;
    float v;
    if      (c < 128)            v = emb[sSrc[m] * DIN + c];
    else if (c < 256)            v = emb[sDst[m] * DIN + (c - 128)];
    else if (c < 256 + DEDGE)    v = edgeF[e * DEDGE + (c - 256)];
    else if (c < KMSG_R)         v = timeF[e * DTIME + (c - 256 - DEDGE)];
    else                         v = 0.0f;
    Am[i] = (_Float16)v;
  }
  __syncthreads();

  // ---- message MLP: GEMM1 (K=544, 17 WMMAs) + bias + ReLU -> Hs ----
  {
    v8f acc = {};
    const _Float16* aRow = &Am[col * KMSG];
    const _Float16* bRow = &W1t[nCol * KMSG];
    for (int kc = 0; kc < KMSG / 32; ++kc)
      acc = wmma16(frag16(aRow, kc * 32, half), frag16(bRow, kc * 32, half), acc);
    float bias = b1[nCol];
    for (int r = 0; r < 8; ++r) {
      float v = acc[r] + bias;
      Hs[(r + 8 * half) * 128 + nCol] = (_Float16)(v > 0.0f ? v : 0.0f);
    }
  }
  __syncthreads();

  // ---- message MLP: GEMM2 (K=128, 4 WMMAs) + bias -> Ms ----
  {
    v8f acc = {};
    const _Float16* aRow = &Hs[col * 128];
    const _Float16* bRow = &W2t[nCol * 128];
    for (int kc = 0; kc < 4; ++kc)
      acc = wmma16(frag16(aRow, kc * 32, half), frag16(bRow, kc * 32, half), acc);
    float bias = b2[nCol];
    for (int r = 0; r < 8; ++r)
      Ms[(r + 8 * half) * 128 + nCol] = (_Float16)(acc[r] + bias);
  }
  __syncthreads();   // all reads of Hs done; safe to reuse below

  // ---- update MLP for side 0 (dst) then side 1 (src) ----
  for (int side = 0; side < 2; ++side) {
    // Hidden: K=256 split as [endpoint emb (from Am) | message (from Ms)]
    v8f h = {};
    const _Float16* eRow = &Am[col * KMSG + (side == 0 ? 128 : 0)];
    const _Float16* mRow = &Ms[col * 128];
    const _Float16* uRow = &U1t[nCol * KUPD];
    for (int kc = 0; kc < 4; ++kc)
      h = wmma16(frag16(eRow, kc * 32, half), frag16(uRow, kc * 32, half), h);
    for (int kc = 0; kc < 4; ++kc)
      h = wmma16(frag16(mRow, kc * 32, half), frag16(uRow, 128 + kc * 32, half), h);
    float bias = ub1[nCol];
    for (int r = 0; r < 8; ++r) {
      float v = h[r] + bias;
      Hs[(r + 8 * half) * 128 + nCol] = (_Float16)(v > 0.0f ? v : 0.0f);
    }
    __syncthreads();

    // Output: K=128, 4 WMMAs; write only rows that won the scatter.
    v8f o = {};
    const _Float16* hRow = &Hs[col * 128];
    const _Float16* bRow = &U2t[nCol * 128];
    for (int kc = 0; kc < 4; ++kc)
      o = wmma16(frag16(hRow, kc * 32, half), frag16(bRow, kc * 32, half), o);
    float ob = ub2[nCol];
    for (int r = 0; r < 8; ++r) {
      int m    = r + 8 * half;
      int e    = e0 + m;
      if (e >= nEdges) continue;
      int node = side ? sSrc[m] : sDst[m];
      int pos  = 2 * e + side;
      if (winner[node] == pos)
        out[node * DIN + nCol] = o[r] + ob;
    }
    __syncthreads();   // before next side reuses Hs
  }
}

// ---------------------------------------------------------------------------
extern "C" void kernel_launch(void* const* d_in, const int* in_sizes, int n_in,
                              void* d_out, int out_size, void* d_ws, size_t ws_size,
                              hipStream_t stream) {
  const float* emb    = (const float*)d_in[0];
  const int*   srcIds = (const int*)d_in[1];
  const int*   dstIds = (const int*)d_in[2];
  const float* edgeF  = (const float*)d_in[3];
  const float* timeF  = (const float*)d_in[4];
  const float* msgW1  = (const float*)d_in[5];
  const float* msgB1  = (const float*)d_in[6];
  const float* msgW2  = (const float*)d_in[7];
  const float* msgB2  = (const float*)d_in[8];
  const float* updW1  = (const float*)d_in[9];
  const float* updB1  = (const float*)d_in[10];
  const float* updW2  = (const float*)d_in[11];
  const float* updB2  = (const float*)d_in[12];
  float* out = (float*)d_out;

  const int nNodes = in_sizes[0] / DIN;
  const int nEdges = in_sizes[1];

  // Workspace layout (winner first keeps 16B alignment for f16 weights)
  char* ws = (char*)d_ws;
  size_t winBytes = ((size_t)nNodes * sizeof(int) + 15) & ~(size_t)15;
  int*      winner = (int*)ws;
  _Float16* w1t = (_Float16*)(ws + winBytes);     // [128][544]
  _Float16* w2t = w1t + 128 * KMSG;               // [128][128]
  _Float16* u1t = w2t + 128 * 128;                // [128][256]
  _Float16* u2t = u1t + 128 * KUPD;               // [128][128]

  prep_weights<<<256, 256, 0, stream>>>(msgW1, msgW2, updW1, updW2,
                                        w1t, w2t, u1t, u2t);

  winner_init<<<(nNodes + 255) / 256, 256, 0, stream>>>(winner, nNodes);
  winner_scatter<<<(nEdges + 255) / 256, 256, 0, stream>>>(srcIds, dstIds,
                                                           winner, nEdges);

  const int n4 = nNodes * DIN / 4;
  copy_base<<<(n4 + 255) / 256, 256, 0, stream>>>((const float4*)emb,
                                                  (float4*)out, n4);

  edge_tile_kernel<<<(nEdges + 15) / 16, 256, 0, stream>>>(
      emb, srcIds, dstIds, edgeF, timeF,
      w1t, msgB1, w2t, msgB2, u1t, updB1, u2t, updB2,
      winner, out, nEdges);
}
